// DepthTracker_17051020165920
// MI455X (gfx1250) — compile-verified, matrix operand order checked by reading, not run on hardware
//
#include <hip/hip_runtime.h>

// Problem constants (from reference)
#define LEVELS 4
#define TT     16
#define TSPLIT 2
#define TCHUNK (TT / TSPLIT)  // 8 timesteps per block
#define CC     128
#define NPTS   256
#define KW     7
#define KK     49      // KW*KW
#define MT     64      // padded tile rows (4 x 16)
#define LDSK   136     // padded K stride in halfs (128 + 8) -> 272B row stride

typedef _Float16 v8h  __attribute__((ext_vector_type(8)));
typedef _Float16 v16h __attribute__((ext_vector_type(16)));
typedef float    v8f  __attribute__((ext_vector_type(8)));

// ---- CDNA5 async global->LDS (bypasses VGPRs, tracked by ASYNCcnt) ----
__device__ __forceinline__ void async_load_b128_to_lds(unsigned lds_byte_off,
                                                       const void* gaddr) {
  asm volatile("global_load_async_to_lds_b128 %0, %1, off"
               :: "v"(lds_byte_off),
                  "v"((unsigned long long)(uintptr_t)gaddr)
               : "memory");
}
__device__ __forceinline__ void wait_asynccnt0() {
  asm volatile("s_wait_asynccnt 0" ::: "memory");
}

// Load a 16x32 (MxK) f16 WMMA fragment row-slice from LDS.
// Wave32 layout: lanes 0-15 hold K = k0+{0..7, 16..23}; lanes 16-31 hold
// K = k0+{8..15, 24..31}; M (or N for B) = lane & 15.
__device__ __forceinline__ v16h load_frag(const _Float16* __restrict__ base,
                                          int row, int k0, int hs) {
  const _Float16* p = base + row * LDSK + k0 + hs * 8;
  v8h lo = *(const v8h*)(p);
  v8h hi = *(const v8h*)(p + 16);
  return __builtin_shufflevector(lo, hi, 0, 1, 2, 3, 4, 5, 6, 7,
                                         8, 9, 10, 11, 12, 13, 14, 15);
}

// Bilinearly sample 49 offsets x 128 channels into an f16 LDS tile.
__device__ __forceinline__ void stage_A(_Float16* __restrict__ dst,
                                        const float* __restrict__ fmt,
                                        float cx, float cy, int Hl, int Wl,
                                        int c, int HW) {
  #pragma unroll 1
  for (int r = 0; r < KK; ++r) {
    // delta flat order: r = i*7 + j, x-offset = i-3, y-offset = j-3
    const float ox = (float)(r / KW - 3);
    const float oy = (float)(r % KW - 3);
    float px = cx + ox;
    float py = cy + oy;
    px = fminf(fmaxf(px, 0.0f), (float)(Wl - 1));
    py = fminf(fmaxf(py, 0.0f), (float)(Hl - 1));
    const float x0f = floorf(px);
    const float y0f = floorf(py);
    const float wx = px - x0f;
    const float wy = py - y0f;
    const int x0 = (int)x0f;
    const int y0 = (int)y0f;
    const int x1 = min(x0 + 1, Wl - 1);
    const int y1 = min(y0 + 1, Hl - 1);
    const float w00 = (1.0f - wy) * (1.0f - wx);
    const float w01 = (1.0f - wy) * wx;
    const float w10 = wy * (1.0f - wx);
    const float w11 = wy * wx;
    const float* pc = fmt + (size_t)c * HW;
    const float v00 = pc[y0 * Wl + x0];
    const float v01 = pc[y0 * Wl + x1];
    const float v10 = pc[y1 * Wl + x0];
    const float v11 = pc[y1 * Wl + x1];
    dst[r * LDSK + c] = (_Float16)(v00 * w00 + v01 * w01 + v10 * w10 + v11 * w11);
  }
}

__global__ __launch_bounds__(128)
void corr_all_kernel(const float* __restrict__ fm0, const float* __restrict__ fm1,
                     const float* __restrict__ fm2, const float* __restrict__ fm3,
                     const float* __restrict__ tr0, const float* __restrict__ tr1,
                     const float* __restrict__ tr2, const float* __restrict__ tr3,
                     const float* __restrict__ coords,
                     float* __restrict__ out) {
  __shared__ _Float16 Blds[MT * LDSK];      // track templates, f16 (N x K)
  __shared__ _Float16 Alds[2][MT * LDSK];   // double-buffered sampled features
  __shared__ float    Bstage[KK * CC];      // fp32 async landing zone (25KB)

  const int n    = blockIdx.x;
  const int lev  = blockIdx.y;
  const int tseg = blockIdx.z;
  const int tid  = threadIdx.x;
  const int wav  = tid >> 5;
  const int lane = tid & 31;
  const int lrow = lane & 15;
  const int hs   = lane >> 4;
  const int c    = tid;                 // channel id for staging

  const float* fm = (lev == 0) ? fm0 : (lev == 1) ? fm1 : (lev == 2) ? fm2 : fm3;
  const float* tr = (lev == 0) ? tr0 : (lev == 1) ? tr1 : (lev == 2) ? tr2 : tr3;
  const int Hl = 96 >> lev;
  const int Wl = 128 >> lev;
  const int HW = Hl * Wl;
  const float scale = 1.0f / (float)(1 << lev);

  // ---- 1) Kick off async DMA of the track slice (fp32) into LDS ----
  // One b128 per lane per row: row r -> 32 lanes x 16B = 512B = 128 floats.
  const unsigned bstage_base = (unsigned)(uintptr_t)Bstage;
  for (int r = wav; r < KK; r += 4) {
    async_load_b128_to_lds(bstage_base + (unsigned)(r * CC + lane * 4) * 4u,
                           tr + ((size_t)r * NPTS + n) * CC + lane * 4);
  }

  // ---- 2) Zero pad rows (rows >= 49 feed only unstored outputs) ----
  for (int r = KK; r < MT; ++r) {
    Blds[r * LDSK + c]    = (_Float16)0.0f;
    Alds[0][r * LDSK + c] = (_Float16)0.0f;
    Alds[1][r * LDSK + c] = (_Float16)0.0f;
  }

  wait_asynccnt0();
  __syncthreads();

  // ---- 3) Convert track fp32 -> f16 WMMA tile (once per block) ----
  #pragma unroll 1
  for (int r = 0; r < KK; ++r) {
    Blds[r * LDSK + c] = (_Float16)Bstage[r * CC + c];
  }

  // ---- 4) Prime first A tile ----
  const int tbase = tseg * TCHUNK;
  {
    const float cx = coords[((size_t)tbase * NPTS + n) * 2 + 0] * scale;
    const float cy = coords[((size_t)tbase * NPTS + n) * 2 + 1] * scale;
    stage_A(Alds[0], fm + (size_t)tbase * CC * HW, cx, cy, Hl, Wl, c, HW);
  }
  __syncthreads();

  // ---- 5) Timestep loop: stage A(t+1) while WMMA'ing A(t) ----
  for (int ti = 0; ti < TCHUNK; ++ti) {
    const int t = tbase + ti;

    if (ti + 1 < TCHUNK) {
      const int t2 = t + 1;
      const float cx2 = coords[((size_t)t2 * NPTS + n) * 2 + 0] * scale;
      const float cy2 = coords[((size_t)t2 * NPTS + n) * 2 + 1] * scale;
      stage_A(Alds[(ti + 1) & 1], fm + (size_t)t2 * CC * HW, cx2, cy2,
              Hl, Wl, c, HW);
    }

    const _Float16* Ab = Alds[ti & 1];
    const int m0 = wav * 16;
    v8f acc0 = {}; v8f acc1 = {}; v8f acc2 = {}; v8f acc3 = {};

    #pragma unroll
    for (int ks = 0; ks < 4; ++ks) {
      const int k0 = ks * 32;
      const v16h a  = load_frag(Ab,   m0 + lrow, k0, hs);
      const v16h b0 = load_frag(Blds,  0 + lrow, k0, hs);
      acc0 = __builtin_amdgcn_wmma_f32_16x16x32_f16(false, a, false, b0,
                                                    (short)0, acc0, false, false);
      const v16h b1 = load_frag(Blds, 16 + lrow, k0, hs);
      acc1 = __builtin_amdgcn_wmma_f32_16x16x32_f16(false, a, false, b1,
                                                    (short)0, acc1, false, false);
      const v16h b2 = load_frag(Blds, 32 + lrow, k0, hs);
      acc2 = __builtin_amdgcn_wmma_f32_16x16x32_f16(false, a, false, b2,
                                                    (short)0, acc2, false, false);
      const v16h b3 = load_frag(Blds, 48 + lrow, k0, hs);
      acc3 = __builtin_amdgcn_wmma_f32_16x16x32_f16(false, a, false, b3,
                                                    (short)0, acc3, false, false);
    }

    // D layout: col = lane&15 (+16*ntile), row = vgpr + 8*hs + m0
    float* outp = out + ((((size_t)lev * TT + t) * NPTS + n) * KK) * KK;
    v8f accs[4] = {acc0, acc1, acc2, acc3};
    #pragma unroll
    for (int nt = 0; nt < 4; ++nt) {
      const int col = nt * 16 + lrow;
      if (col < KK) {
        #pragma unroll
        for (int v = 0; v < 8; ++v) {
          const int m = m0 + v + 8 * hs;
          if (m < KK) {
            outp[(size_t)m * KK + col] = accs[nt][v];
          }
        }
      }
    }

    __syncthreads();
  }
}

extern "C" void kernel_launch(void* const* d_in, const int* in_sizes, int n_in,
                              void* d_out, int out_size, void* d_ws, size_t ws_size,
                              hipStream_t stream) {
  (void)in_sizes; (void)n_in; (void)out_size; (void)d_ws; (void)ws_size;
  const float* coords = (const float*)d_in[8];
  float* out = (float*)d_out;

  dim3 grid(NPTS, LEVELS, TSPLIT);
  corr_all_kernel<<<grid, 128, 0, stream>>>(
      (const float*)d_in[0], (const float*)d_in[1],
      (const float*)d_in[2], (const float*)d_in[3],
      (const float*)d_in[4], (const float*)d_in[5],
      (const float*)d_in[6], (const float*)d_in[7],
      coords, out);
}
